// get_model_76450417869450
// MI455X (gfx1250) — compile-verified, matrix operand order checked by reading, not run on hardware
//
#include <hip/hip_runtime.h>
#include <hip/hip_bf16.h>

// ---------------------------------------------------------------------------
// Types for CDNA5 WMMA
// ---------------------------------------------------------------------------
typedef _Float16 v16h __attribute__((ext_vector_type(16)));
typedef float    v8f  __attribute__((ext_vector_type(8)));
typedef _Float16 h2   __attribute__((ext_vector_type(2)));

#define BBALL 16      // total clouds (12 support + 4 query)
#define NGRP0 12      // clouds in BN group 0 (support backbone call)

// ---------------------------------------------------------------------------
// Farthest point sampling: one block per cloud, iterative (matches jax scan:
// idx[0]=0, then argmax of running min-distance, first-index tie break).
// ---------------------------------------------------------------------------
__global__ __launch_bounds__(256)
void fps_kernel(const float* __restrict__ xyz, int* __restrict__ idx,
                int N, int npoint)
{
    __shared__ float dist[2048];
    __shared__ float rb[256];
    __shared__ int   ri[256];
    __shared__ int   ssel;
    const int bb = blockIdx.x;
    const float* base = xyz + (long long)bb * N * 3;
    for (int n = threadIdx.x; n < N; n += 256) dist[n] = 1e10f;
    if (threadIdx.x == 0) ssel = 0;
    __syncthreads();
    for (int it = 0; it < npoint; ++it) {
        int sel = ssel;
        if (threadIdx.x == 0) idx[(long long)bb * npoint + it] = sel;
        float cx = base[sel * 3 + 0], cy = base[sel * 3 + 1], cz = base[sel * 3 + 2];
        float best = -1.f; int bi = 0x7fffffff;
        for (int n = threadIdx.x; n < N; n += 256) {
            float dx = base[n * 3 + 0] - cx;
            float dy = base[n * 3 + 1] - cy;
            float dz = base[n * 3 + 2] - cz;
            float d  = dx * dx + dy * dy + dz * dz;
            float nd = fminf(dist[n], d);
            dist[n] = nd;
            if (nd > best) { best = nd; bi = n; }
        }
        rb[threadIdx.x] = best; ri[threadIdx.x] = bi;
        __syncthreads();
        for (int o = 128; o > 0; o >>= 1) {
            if (threadIdx.x < o) {
                float ob = rb[threadIdx.x + o]; int oi = ri[threadIdx.x + o];
                if (ob > rb[threadIdx.x] ||
                    (ob == rb[threadIdx.x] && oi < ri[threadIdx.x])) {
                    rb[threadIdx.x] = ob; ri[threadIdx.x] = oi;
                }
            }
            __syncthreads();
        }
        if (threadIdx.x == 0) ssel = ri[0];
        __syncthreads();
    }
}

__global__ void gather_xyz_kernel(const float* __restrict__ xyz,
                                  const int* __restrict__ idx,
                                  float* __restrict__ out, int N, int npoint)
{
    long long i = (long long)blockIdx.x * blockDim.x + threadIdx.x;
    long long total = (long long)BBALL * npoint;
    if (i >= total) return;
    int bb = (int)(i / npoint);
    int g  = idx[i];
    const float* p = xyz + ((long long)bb * N + g) * 3;
    out[i * 3 + 0] = p[0]; out[i * 3 + 1] = p[1]; out[i * 3 + 2] = p[2];
}

// Ball query == first nsample in-range indices in ascending order, padded with
// the first hit (equivalent to the reference's where/sort/slice construction).
__global__ void query_ball_kernel(const float* __restrict__ xyz,
                                  const float* __restrict__ nxyz,
                                  int* __restrict__ gidx,
                                  int N, int npoint, int ns, float r2)
{
    long long i = (long long)blockIdx.x * blockDim.x + threadIdx.x;
    long long total = (long long)BBALL * npoint;
    if (i >= total) return;
    int bb = (int)(i / npoint);
    const float* base = xyz + (long long)bb * N * 3;
    float cx = nxyz[i * 3 + 0], cy = nxyz[i * 3 + 1], cz = nxyz[i * 3 + 2];
    int* out = gidx + i * ns;
    int cnt = 0, first = 0;
    for (int k = 0; k < N && cnt < ns; ++k) {
        float dx = base[k * 3 + 0] - cx;
        float dy = base[k * 3 + 1] - cy;
        float dz = base[k * 3 + 2] - cz;
        float d = dx * dx + dy * dy + dz * dz;
        if (d <= r2) { if (cnt == 0) first = k; out[cnt++] = k; }
    }
    for (; cnt < ns; ++cnt) out[cnt] = first;
}

// 3-NN (smallest distances, stable tie order) + inverse-distance weights.
__global__ void knn3_kernel(const float* __restrict__ q,  // [BB][Nq][3]
                            const float* __restrict__ s,  // [BB][Ns][3]
                            int* __restrict__ oidx, float* __restrict__ ow,
                            int Nq, int Ns)
{
    long long i = (long long)blockIdx.x * blockDim.x + threadIdx.x;
    long long total = (long long)BBALL * Nq;
    if (i >= total) return;
    int bb = (int)(i / Nq);
    float qx = q[i * 3 + 0], qy = q[i * 3 + 1], qz = q[i * 3 + 2];
    const float* base = s + (long long)bb * Ns * 3;
    float d0 = 1e30f, d1 = 1e30f, d2 = 1e30f;
    int i0 = 0, i1 = 0, i2 = 0;
    for (int k = 0; k < Ns; ++k) {
        float dx = base[k * 3 + 0] - qx;
        float dy = base[k * 3 + 1] - qy;
        float dz = base[k * 3 + 2] - qz;
        float d = dx * dx + dy * dy + dz * dz;
        if (d < d0)      { d2 = d1; i2 = i1; d1 = d0; i1 = i0; d0 = d; i0 = k; }
        else if (d < d1) { d2 = d1; i2 = i1; d1 = d;  i1 = k; }
        else if (d < d2) { d2 = d;  i2 = k; }
    }
    float r0 = 1.f / (d0 + 1e-8f), r1 = 1.f / (d1 + 1e-8f), r2v = 1.f / (d2 + 1e-8f);
    float rs = r0 + r1 + r2v;
    oidx[i * 3 + 0] = i0; oidx[i * 3 + 1] = i1; oidx[i * 3 + 2] = i2;
    ow[i * 3 + 0] = r0 / rs; ow[i * 3 + 1] = r1 / rs; ow[i * 3 + 2] = r2v / rs;
}

// ---------------------------------------------------------------------------
// Grouped-feature builders -> packed f16 activations X[(bb*Mper+m)][Kpad]
// ---------------------------------------------------------------------------
__global__ void build_sa1_kernel(const float* __restrict__ xyz,
                                 const float* __restrict__ nxyz,
                                 const int* __restrict__ gidx,
                                 _Float16* __restrict__ X,
                                 int N, int npoint, int ns)  // Kpad = 32
{
    long long M = (long long)npoint * ns;
    long long i = (long long)blockIdx.x * blockDim.x + threadIdx.x;
    if (i >= (long long)BBALL * M) return;
    int bb = (int)(i / M);
    long long m = i % M;
    int n = (int)(m / ns), sI = (int)(m % ns);
    int g = gidx[((long long)bb * npoint + n) * ns + sI];
    const float* p = xyz  + ((long long)bb * N + g) * 3;
    const float* c = nxyz + ((long long)bb * npoint + n) * 3;
    _Float16* o = X + i * 32;
    o[0] = (_Float16)(p[0] - c[0]);
    o[1] = (_Float16)(p[1] - c[1]);
    o[2] = (_Float16)(p[2] - c[2]);
    o[3] = (_Float16)p[0]; o[4] = (_Float16)p[1]; o[5] = (_Float16)p[2];
    for (int k = 6; k < 32; ++k) o[k] = (_Float16)0.f;
}

__global__ void build_sa2_kernel(const float* __restrict__ xyz,
                                 const float* __restrict__ nxyz,
                                 const int* __restrict__ gidx,
                                 const float* __restrict__ pts, // [BB][Cp][N]
                                 _Float16* __restrict__ X,
                                 int N, int npoint, int ns, int Cp, int Kpad)
{
    long long M = (long long)npoint * ns;
    long long i = (long long)blockIdx.x * blockDim.x + threadIdx.x;
    if (i >= (long long)BBALL * M) return;
    int bb = (int)(i / M);
    long long m = i % M;
    int n = (int)(m / ns), sI = (int)(m % ns);
    int g = gidx[((long long)bb * npoint + n) * ns + sI];
    const float* p = xyz  + ((long long)bb * N + g) * 3;
    const float* c = nxyz + ((long long)bb * npoint + n) * 3;
    _Float16* o = X + i * Kpad;
    o[0] = (_Float16)(p[0] - c[0]);
    o[1] = (_Float16)(p[1] - c[1]);
    o[2] = (_Float16)(p[2] - c[2]);
    const float* pb = pts + (long long)bb * Cp * N + g;
    for (int cch = 0; cch < Cp; ++cch) o[3 + cch] = (_Float16)pb[(long long)cch * N];
    for (int k = 3 + Cp; k < Kpad; ++k) o[k] = (_Float16)0.f;
}

__global__ void build_sa3_kernel(const float* __restrict__ xyz2, // [BB][Np][3]
                                 const float* __restrict__ pts,  // [BB][Cp][Np]
                                 _Float16* __restrict__ X,
                                 int Np, int Cp, int Kpad)
{
    long long i = (long long)blockIdx.x * blockDim.x + threadIdx.x;
    if (i >= (long long)BBALL * Np) return;
    int bb = (int)(i / Np);
    int m  = (int)(i % Np);
    _Float16* o = X + i * Kpad;
    o[0] = (_Float16)xyz2[i * 3 + 0];
    o[1] = (_Float16)xyz2[i * 3 + 1];
    o[2] = (_Float16)xyz2[i * 3 + 2];
    const float* pb = pts + (long long)bb * Cp * Np + m;
    for (int cch = 0; cch < Cp; ++cch) o[3 + cch] = (_Float16)pb[(long long)cch * Np];
    for (int k = 3 + Cp; k < Kpad; ++k) o[k] = (_Float16)0.f;
}

__global__ void build_fp3_kernel(const float* __restrict__ l2pts, // [BB][256][128]
                                 const float* __restrict__ l3pts, // [BB][1024]
                                 _Float16* __restrict__ X)        // Kpad=1280
{
    long long i = (long long)blockIdx.x * blockDim.x + threadIdx.x;
    if (i >= (long long)BBALL * 128) return;
    int bb = (int)(i / 128);
    int m  = (int)(i % 128);
    _Float16* o = X + i * 1280;
    const float* a = l2pts + (long long)bb * 256 * 128 + m;
    for (int c = 0; c < 256; ++c) o[c] = (_Float16)a[(long long)c * 128];
    const float* b = l3pts + (long long)bb * 1024;
    for (int c = 0; c < 1024; ++c) o[256 + c] = (_Float16)b[c];
}

__global__ void build_fp2_kernel(const float* __restrict__ l1pts, // [BB][128][512]
                                 const float* __restrict__ l2fp,  // [BB][256][128]
                                 const int* __restrict__ kidx,    // [BB*512][3]
                                 const float* __restrict__ kw,
                                 _Float16* __restrict__ X)        // Kpad=384
{
    long long i = (long long)blockIdx.x * blockDim.x + threadIdx.x;
    if (i >= (long long)BBALL * 512) return;
    int bb = (int)(i / 512);
    int m  = (int)(i % 512);
    _Float16* o = X + i * 384;
    const float* a = l1pts + (long long)bb * 128 * 512 + m;
    for (int c = 0; c < 128; ++c) o[c] = (_Float16)a[(long long)c * 512];
    int i0 = kidx[i * 3 + 0], i1 = kidx[i * 3 + 1], i2 = kidx[i * 3 + 2];
    float w0 = kw[i * 3 + 0], w1 = kw[i * 3 + 1], w2 = kw[i * 3 + 2];
    const float* b = l2fp + (long long)bb * 256 * 128;
    for (int c = 0; c < 256; ++c) {
        const float* r = b + (long long)c * 128;
        o[128 + c] = (_Float16)(w0 * r[i0] + w1 * r[i1] + w2 * r[i2]);
    }
}

__global__ void build_fp1_kernel(const float* __restrict__ xyz0, // [BB][2048][3]
                                 const float* __restrict__ l1fp, // [BB][128][512]
                                 const int* __restrict__ kidx,   // [BB*2048][3]
                                 const float* __restrict__ kw,
                                 _Float16* __restrict__ X)       // Kpad=160
{
    long long i = (long long)blockIdx.x * blockDim.x + threadIdx.x;
    if (i >= (long long)BBALL * 2048) return;
    int bb = (int)(i / 2048);
    _Float16* o = X + i * 160;
    float px = xyz0[i * 3 + 0], py = xyz0[i * 3 + 1], pz = xyz0[i * 3 + 2];
    o[0] = (_Float16)px; o[1] = (_Float16)py; o[2] = (_Float16)pz;
    o[3] = (_Float16)px; o[4] = (_Float16)py; o[5] = (_Float16)pz;
    int i0 = kidx[i * 3 + 0], i1 = kidx[i * 3 + 1], i2 = kidx[i * 3 + 2];
    float w0 = kw[i * 3 + 0], w1 = kw[i * 3 + 1], w2 = kw[i * 3 + 2];
    const float* b = l1fp + (long long)bb * 128 * 512;
    for (int c = 0; c < 128; ++c) {
        const float* r = b + (long long)c * 512;
        o[6 + c] = (_Float16)(w0 * r[i0] + w1 * r[i1] + w2 * r[i2]);
    }
    for (int k = 134; k < 160; ++k) o[k] = (_Float16)0.f;
}

// ---------------------------------------------------------------------------
// Weight f32 -> f16 with K zero-padding:  W16[Cout][Kpad]
// ---------------------------------------------------------------------------
__global__ void convert_w_kernel(const float* __restrict__ W,
                                 _Float16* __restrict__ W16,
                                 int Cin, int Cout, int Kpad)
{
    long long i = (long long)blockIdx.x * blockDim.x + threadIdx.x;
    if (i >= (long long)Cout * Kpad) return;
    int co = (int)(i / Kpad), k = (int)(i % Kpad);
    W16[i] = (k < Cin) ? (_Float16)W[(long long)co * Cin + k] : (_Float16)0.f;
}

// ---------------------------------------------------------------------------
// WMMA GEMM:  Y[bb][Cout][Mper] = W16[Cout][Kpad] x X16[Mtot][Kpad]^T
// Register-blocked 2x2: each wave owns a 32(C) x 32(M) output block = four
// 16x16 WMMA tiles; per K-step of 32 it loads 2 A-fragments + 2 B-fragments
// and issues 4x v_wmma_f32_16x16x32_f16 (double the data reuse vs 1x1).
// Fragment layout (16-bit 16x32): lane L (half=L>>4, row/col=L&15); VGPR r
// holds the K pair at kbase = (r>>2)*16 + half*8 + (r&3)*2.
// D tile: lane covers N=L&15, VGPR r -> M = r + 8*half.
// All Cout and Mper values in this network are multiples of 32.
// ---------------------------------------------------------------------------
__global__ __launch_bounds__(256)
void wmma_gemm_kernel(const _Float16* __restrict__ W16,
                      const _Float16* __restrict__ X16,
                      float* __restrict__ Y,
                      int Cout, int Kpad, int Mper, int BB)
{
    const int lane = threadIdx.x & 31;
    const int wave = threadIdx.x >> 5;
    const int c0   = blockIdx.x * 32;                              // 2 C tiles
    const long long Mtot = (long long)Mper * BB;
    const long long mt0  = ((long long)blockIdx.y * 8 + wave) * 32; // 2 M tiles
    if (mt0 >= Mtot) return;
    const int half = lane >> 4;
    const int l15  = lane & 15;

    v8f acc00 = {}, acc01 = {}, acc10 = {}, acc11 = {};
    const _Float16* arow0 = W16 + (long long)(c0 + l15) * Kpad;
    const _Float16* arow1 = arow0 + (long long)16 * Kpad;
    const _Float16* brow0 = X16 + (mt0 + l15) * (long long)Kpad;
    const _Float16* brow1 = brow0 + (long long)16 * Kpad;

    for (int k0 = 0; k0 < Kpad; k0 += 32) {
        __builtin_prefetch(brow0 + k0 + 64, 0, 1);   // global_prefetch_b8
        __builtin_prefetch(brow1 + k0 + 64, 0, 1);
        v16h a0, a1, b0, b1;
#pragma unroll
        for (int r = 0; r < 8; ++r) {
            const int kb = k0 + ((r >> 2) << 4) + (half << 3) + ((r & 3) << 1);
            h2 w0 = *(const h2*)(arow0 + kb);
            h2 w1 = *(const h2*)(arow1 + kb);
            h2 x0 = *(const h2*)(brow0 + kb);
            h2 x1 = *(const h2*)(brow1 + kb);
            a0[2 * r] = w0[0]; a0[2 * r + 1] = w0[1];
            a1[2 * r] = w1[0]; a1[2 * r + 1] = w1[1];
            b0[2 * r] = x0[0]; b0[2 * r + 1] = x0[1];
            b1[2 * r] = x1[0]; b1[2 * r + 1] = x1[1];
        }
        acc00 = __builtin_amdgcn_wmma_f32_16x16x32_f16(false, a0, false, b0,
                                                       (short)0, acc00, false, false);
        acc01 = __builtin_amdgcn_wmma_f32_16x16x32_f16(false, a0, false, b1,
                                                       (short)0, acc01, false, false);
        acc10 = __builtin_amdgcn_wmma_f32_16x16x32_f16(false, a1, false, b0,
                                                       (short)0, acc10, false, false);
        acc11 = __builtin_amdgcn_wmma_f32_16x16x32_f16(false, a1, false, b1,
                                                       (short)0, acc11, false, false);
    }

    // Store the four tiles. Output is channel-major per cloud: Y[bb][c][m].
#pragma unroll
    for (int t = 0; t < 2; ++t) {
        const long long mg = mt0 + t * 16 + l15;
        const int bb = (int)(mg / Mper);
        const int ml = (int)(mg % Mper);
        float* ybase = Y + ((long long)bb * Cout) * Mper + ml;
        const v8f accA = t ? acc01 : acc00;   // C rows c0 .. c0+15
        const v8f accB = t ? acc11 : acc10;   // C rows c0+16 .. c0+31
#pragma unroll
        for (int r = 0; r < 8; ++r) {
            const int crow = r + (half << 3);
            ybase[(long long)(c0 + crow) * Mper]      = accA[r];
            ybase[(long long)(c0 + 16 + crow) * Mper] = accB[r];
        }
    }
}

// ---------------------------------------------------------------------------
// BatchNorm stats per (group, channel) over the group's clouds x spatial.
// ---------------------------------------------------------------------------
__global__ __launch_bounds__(256)
void bn_stats_kernel(const float* __restrict__ Y, float* __restrict__ mean,
                     float* __restrict__ var, int Cout, int Mper,
                     int ngroup0, int BB)
{
    const int c = blockIdx.x;
    const int g = blockIdx.y;
    const int bb0 = g ? ngroup0 : 0;
    const int bb1 = g ? BB : ngroup0;
    float s = 0.f, ss = 0.f;
    for (int bb = bb0; bb < bb1; ++bb) {
        const float* p = Y + ((long long)bb * Cout + c) * Mper;
        for (int m = threadIdx.x; m < Mper; m += 256) {
            float v = p[m]; s += v; ss += v * v;
        }
    }
    __shared__ float rs[256], rss[256];
    rs[threadIdx.x] = s; rss[threadIdx.x] = ss;
    __syncthreads();
    for (int o = 128; o > 0; o >>= 1) {
        if (threadIdx.x < o) {
            rs[threadIdx.x]  += rs[threadIdx.x + o];
            rss[threadIdx.x] += rss[threadIdx.x + o];
        }
        __syncthreads();
    }
    if (threadIdx.x == 0) {
        float cnt = (float)(bb1 - bb0) * (float)Mper;
        float mu = rs[0] / cnt;
        mean[g * Cout + c] = mu;
        var[g * Cout + c]  = fmaxf(rss[0] / cnt - mu * mu, 0.f);
    }
}

// BN + ReLU; writes f32 [bb][C][M] (optional) and packed f16 [bb*M][KpadNext]
// (optional, pad channels zeroed).  Bias `b` cancels under BN and is skipped.
__global__ void bn_apply_kernel(const float* __restrict__ Y,
                                const float* __restrict__ mean,
                                const float* __restrict__ var,
                                const float* __restrict__ gamma,
                                const float* __restrict__ beta,
                                float* __restrict__ outF32,
                                _Float16* __restrict__ outF16,
                                int Cout, int KpadNext, int Mper,
                                int BB, int ngroup0)
{
    long long i = (long long)blockIdx.x * blockDim.x + threadIdx.x;
    long long total = (long long)BB * KpadNext * Mper;
    if (i >= total) return;
    int m  = (int)(i % Mper);
    int c  = (int)((i / Mper) % KpadNext);
    int bb = (int)(i / ((long long)Mper * KpadNext));
    float v = 0.f;
    if (c < Cout) {
        int g = (bb >= ngroup0) ? 1 : 0;
        float y = Y[((long long)bb * Cout + c) * Mper + m];
        v = (y - mean[g * Cout + c]) * rsqrtf(var[g * Cout + c] + 1e-5f)
            * gamma[c] + beta[c];
        v = fmaxf(v, 0.f);
        if (outF32) outF32[((long long)bb * Cout + c) * Mper + m] = v;
    }
    if (outF16) outF16[((long long)bb * Mper + m) * KpadNext + c] = (_Float16)v;
}

// Max over the nsample axis: X[bb][C][npoint*ns] -> O[bb][C][npoint]
__global__ void maxpool_kernel(const float* __restrict__ X, float* __restrict__ O,
                               int C, int npoint, int ns)
{
    long long i = (long long)blockIdx.x * blockDim.x + threadIdx.x;
    long long total = (long long)BBALL * C * npoint;
    if (i >= total) return;
    int n  = (int)(i % npoint);
    int c  = (int)((i / npoint) % C);
    int bb = (int)(i / ((long long)npoint * C));
    const float* p = X + ((long long)bb * C + c) * ((long long)npoint * ns)
                       + (long long)n * ns;
    float mx = -3.4e38f;
    for (int s = 0; s < ns; ++s) mx = fmaxf(mx, p[s]);
    O[((long long)bb * C + c) * npoint + n] = mx;
}

// ---------------------------------------------------------------------------
// Few-shot head
// ---------------------------------------------------------------------------
__global__ __launch_bounds__(128)
void proto_sums_kernel(const float* __restrict__ feat, // clouds 0..11 [cl][C][N]
                       const int* __restrict__ sy,     // [cl][N]
                       float* __restrict__ sums, float* __restrict__ counts,
                       int C, int N, int P)
{
    int blk = blockIdx.x;           // cloud*P + p
    int cloud = blk / P, p = blk % P;
    int c = threadIdx.x;
    const float* f = feat + ((long long)cloud * C + c) * N;
    const int* lab = sy + (long long)cloud * N;
    float s = 0.f; int cnt = 0;
    for (int n = 0; n < N; ++n)
        if (lab[n] == p) { s += f[n]; ++cnt; }
    sums[(long long)blk * C + c] = s;
    if (c == 0) counts[blk] = (float)cnt;
}

__global__ __launch_bounds__(128)
void proto_combine_kernel(const float* __restrict__ sums,
                          const float* __restrict__ counts,
                          float* __restrict__ proto, int C, int P, int Shot)
{
    int b = blockIdx.x, p = blockIdx.y, c = threadIdx.x;
    float acc = 0.f, hs = 0.f;
    for (int sh = 0; sh < Shot; ++sh) {
        int blk = (b * Shot + sh) * P + p;
        float cnt = counts[blk];
        float has = cnt > 1.f ? 1.f : 0.f;
        acc += has * sums[(long long)blk * C + c] / fmaxf(cnt, 1.f);
        hs  += has;
    }
    proto[((long long)b * P + p) * C + c] = acc / fmaxf(hs, 1.f);
}

__global__ void euc_softmax_kernel(const float* __restrict__ proto, // [4][P][C]
                                   const float* __restrict__ qfeat, // [4][C][N]
                                   float* __restrict__ out,         // [4][N][P]
                                   int C, int N, int P)
{
    long long i = (long long)blockIdx.x * blockDim.x + threadIdx.x;
    if (i >= (long long)4 * N) return;
    int b = (int)(i / N), n = (int)(i % N);
    float e[8];
    for (int p = 0; p < P; ++p) {
        const float* pr = proto + ((long long)b * P + p) * C;
        const float* qf = qfeat + (long long)b * C * N + n;
        float s = 0.f;
        for (int c = 0; c < C; ++c) {
            float d = pr[c] - qf[(long long)c * N];
            s += d * d;
        }
        e[p] = s / (float)C;
    }
    float mx = e[0];
    for (int p = 1; p < P; ++p) mx = fmaxf(mx, e[p]);
    float sum = 0.f;
    for (int p = 0; p < P; ++p) { e[p] = __expf(e[p] - mx); sum += e[p]; }
    for (int p = 0; p < P; ++p) out[i * P + p] = e[p] / sum;
}

// ---------------------------------------------------------------------------
// Host side
// ---------------------------------------------------------------------------
namespace {

struct Bump {
    char* base; size_t off; size_t cap;
    void* alloc(size_t bytes) {
        size_t a = (off + 255) & ~(size_t)255;
        off = a + bytes;
        return base + a;
    }
    float*    f32(size_t n) { return (float*)alloc(n * 4); }
    _Float16* f16(size_t n) { return (_Float16*)alloc(n * 2); }
    int*      i32(size_t n) { return (int*)alloc(n * 4); }
};

struct ConvP {
    const float* W; const float* g; const float* bt;
    int Cin, Cout, Kpad; _Float16* W16;
};

inline void run_conv(hipStream_t st, const ConvP& L, const _Float16* Xin,
                     float* Yar, float* meanb, float* varb,
                     float* outF32, _Float16* outF16, int KpadNext,
                     int Mper, int BB, int ng0)
{
    {
        long long t = (long long)L.Cout * L.Kpad;
        convert_w_kernel<<<(unsigned)((t + 255) / 256), 256, 0, st>>>(
            L.W, L.W16, L.Cin, L.Cout, L.Kpad);
    }
    long long Mtot = (long long)Mper * BB;
    // 2x2 register-blocked: block = 8 waves * 32 cols = 256 cols, 32 rows.
    dim3 g1((unsigned)(L.Cout / 32), (unsigned)((Mtot + 255) / 256));
    wmma_gemm_kernel<<<g1, 256, 0, st>>>(L.W16, Xin, Yar, L.Cout, L.Kpad, Mper, BB);
    dim3 g2((unsigned)L.Cout, 2);
    bn_stats_kernel<<<g2, 256, 0, st>>>(Yar, meanb, varb, L.Cout, Mper, ng0, BB);
    int kn = outF16 ? KpadNext : L.Cout;
    long long t = (long long)BB * kn * Mper;
    bn_apply_kernel<<<(unsigned)((t + 255) / 256), 256, 0, st>>>(
        Yar, meanb, varb, L.g, L.bt, outF32, outF16, L.Cout, kn, Mper, BB, ng0);
}

} // namespace

extern "C" void kernel_launch(void* const* d_in, const int* in_sizes, int n_in,
                              void* d_out, int out_size, void* d_ws, size_t ws_size,
                              hipStream_t stream)
{
    (void)in_sizes; (void)n_in; (void)out_size;
    const float* sx = (const float*)d_in[0];   // (4,3,2048,3) == 12 clouds
    const int*   sy = (const int*)d_in[1];     // (4,3,2048)
    const float* qx = (const float*)d_in[2];   // (4,2048,3)
    // d_in[3] = num_parts (==6, hardcoded)
    // params leaves (pytree: dict keys sorted -> fp1,fp2,fp3,sa1,sa2,sa3;
    // per-layer keys sorted -> W,b,bt,g), starting at index 4.
    // Layer linear index li: fp1(0..2) fp2(3..4) fp3(5..6) sa1(7..9) sa2(10..12) sa3(13..15)
    static const int CinT[16]  = {134,128,128, 384,256, 1280,256, 6,64,64, 131,128,128, 259,256,512};
    static const int CoutT[16] = {128,128,128, 256,128,  256,256, 64,64,128, 128,128,256, 256,512,1024};

    Bump ws{(char*)d_ws, 0, ws_size};

    // --- persistent buffers ---
    float* xyz0  = ws.f32((size_t)BBALL * 2048 * 3);
    int*   idx1  = ws.i32((size_t)BBALL * 512);
    float* nxyz1 = ws.f32((size_t)BBALL * 512 * 3);
    int*   gidx1 = ws.i32((size_t)BBALL * 512 * 32);
    int*   idx2  = ws.i32((size_t)BBALL * 128);
    float* nxyz2 = ws.f32((size_t)BBALL * 128 * 3);
    int*   gidx2 = ws.i32((size_t)BBALL * 128 * 64);
    float* l1pts = ws.f32((size_t)BBALL * 128 * 512);
    float* l2pts = ws.f32((size_t)BBALL * 256 * 128);
    float* l3pts = ws.f32((size_t)BBALL * 1024);
    float* l2fp  = ws.f32((size_t)BBALL * 256 * 128);
    float* l1fp  = ws.f32((size_t)BBALL * 128 * 512);
    float* featF = ws.f32((size_t)BBALL * 128 * 2048);
    int*   kidx2 = ws.i32((size_t)BBALL * 512 * 3);
    float* kw2   = ws.f32((size_t)BBALL * 512 * 3);
    int*   kidx1 = ws.i32((size_t)BBALL * 2048 * 3);
    float* kw1   = ws.f32((size_t)BBALL * 2048 * 3);
    float* meanb = ws.f32(2 * 1024);
    float* varb  = ws.f32(2 * 1024);
    float* psums = ws.f32((size_t)12 * 6 * 128);
    float* pcnts = ws.f32((size_t)12 * 6);
    float* proto = ws.f32((size_t)4 * 6 * 128);

    ConvP L[16];
    for (int li = 0; li < 16; ++li) {
        L[li].W  = (const float*)d_in[4 + li * 4 + 0];
        L[li].bt = (const float*)d_in[4 + li * 4 + 2];
        L[li].g  = (const float*)d_in[4 + li * 4 + 3];
        L[li].Cin = CinT[li]; L[li].Cout = CoutT[li];
        L[li].Kpad = (CinT[li] + 31) / 32 * 32;
        L[li].W16 = ws.f16((size_t)L[li].Cout * L[li].Kpad);
    }

    // --- arenas ---
    float*    Yar = ws.f32(33554432);          // max Cout*M*BB = 128*16384*16
    float*    Nrm = ws.f32(33554432);
    _Float16* FA  = ws.f16(22020096);          // max M*Kpad*BB = 8192*160*16
    _Float16* FB  = ws.f16(22020096);

    // --- assemble xyz0 = [12 support clouds ; 4 query clouds] ---
    hipMemcpyAsync(xyz0, sx, (size_t)12 * 2048 * 3 * 4,
                   hipMemcpyDeviceToDevice, stream);
    hipMemcpyAsync(xyz0 + (size_t)12 * 2048 * 3, qx, (size_t)4 * 2048 * 3 * 4,
                   hipMemcpyDeviceToDevice, stream);

    auto blk = [](long long t) { return (unsigned)((t + 255) / 256); };

    // ================= SA1 =================
    fps_kernel<<<BBALL, 256, 0, stream>>>(xyz0, idx1, 2048, 512);
    gather_xyz_kernel<<<blk((long long)BBALL * 512), 256, 0, stream>>>(
        xyz0, idx1, nxyz1, 2048, 512);
    query_ball_kernel<<<blk((long long)BBALL * 512), 256, 0, stream>>>(
        xyz0, nxyz1, gidx1, 2048, 512, 32, 0.04f);
    build_sa1_kernel<<<blk((long long)BBALL * 16384), 256, 0, stream>>>(
        xyz0, nxyz1, gidx1, FA, 2048, 512, 32);
    run_conv(stream, L[7], FA, Yar, meanb, varb, nullptr, FB, 64, 16384, BBALL, NGRP0);
    run_conv(stream, L[8], FB, Yar, meanb, varb, nullptr, FA, 64, 16384, BBALL, NGRP0);
    run_conv(stream, L[9], FA, Yar, meanb, varb, Nrm, nullptr, 0, 16384, BBALL, NGRP0);
    maxpool_kernel<<<blk((long long)BBALL * 128 * 512), 256, 0, stream>>>(
        Nrm, l1pts, 128, 512, 32);

    // ================= SA2 =================
    fps_kernel<<<BBALL, 256, 0, stream>>>(nxyz1, idx2, 512, 128);
    gather_xyz_kernel<<<blk((long long)BBALL * 128), 256, 0, stream>>>(
        nxyz1, idx2, nxyz2, 512, 128);
    query_ball_kernel<<<blk((long long)BBALL * 128), 256, 0, stream>>>(
        nxyz1, nxyz2, gidx2, 512, 128, 64, 0.16f);
    build_sa2_kernel<<<blk((long long)BBALL * 8192), 256, 0, stream>>>(
        nxyz1, nxyz2, gidx2, l1pts, FA, 512, 128, 64, 128, 160);
    run_conv(stream, L[10], FA, Yar, meanb, varb, nullptr, FB, 128, 8192, BBALL, NGRP0);
    run_conv(stream, L[11], FB, Yar, meanb, varb, nullptr, FA, 128, 8192, BBALL, NGRP0);
    run_conv(stream, L[12], FA, Yar, meanb, varb, Nrm, nullptr, 0, 8192, BBALL, NGRP0);
    maxpool_kernel<<<blk((long long)BBALL * 256 * 128), 256, 0, stream>>>(
        Nrm, l2pts, 256, 128, 64);

    // ================= SA3 (group_all) =================
    build_sa3_kernel<<<blk((long long)BBALL * 128), 256, 0, stream>>>(
        nxyz2, l2pts, FA, 128, 256, 288);
    run_conv(stream, L[13], FA, Yar, meanb, varb, nullptr, FB, 256, 128, BBALL, NGRP0);
    run_conv(stream, L[14], FB, Yar, meanb, varb, nullptr, FA, 512, 128, BBALL, NGRP0);
    run_conv(stream, L[15], FA, Yar, meanb, varb, Nrm, nullptr, 0, 128, BBALL, NGRP0);
    maxpool_kernel<<<blk((long long)BBALL * 1024), 256, 0, stream>>>(
        Nrm, l3pts, 1024, 1, 128);

    // ================= FP3 (broadcast interp) =================
    build_fp3_kernel<<<blk((long long)BBALL * 128), 256, 0, stream>>>(l2pts, l3pts, FA);
    run_conv(stream, L[5], FA, Yar, meanb, varb, nullptr, FB, 256, 128, BBALL, NGRP0);
    run_conv(stream, L[6], FB, Yar, meanb, varb, l2fp, nullptr, 0, 128, BBALL, NGRP0);

    // ================= FP2 (3-NN interp) =================
    knn3_kernel<<<blk((long long)BBALL * 512), 256, 0, stream>>>(
        nxyz1, nxyz2, kidx2, kw2, 512, 128);
    build_fp2_kernel<<<blk((long long)BBALL * 512), 256, 0, stream>>>(
        l1pts, l2fp, kidx2, kw2, FA);
    run_conv(stream, L[3], FA, Yar, meanb, varb, nullptr, FB, 256, 512, BBALL, NGRP0);
    run_conv(stream, L[4], FB, Yar, meanb, varb, l1fp, nullptr, 0, 512, BBALL, NGRP0);

    // ================= FP1 =================
    knn3_kernel<<<blk((long long)BBALL * 2048), 256, 0, stream>>>(
        xyz0, nxyz1, kidx1, kw1, 2048, 512);
    build_fp1_kernel<<<blk((long long)BBALL * 2048), 256, 0, stream>>>(
        xyz0, l1fp, kidx1, kw1, FA);
    run_conv(stream, L[0], FA, Yar, meanb, varb, nullptr, FB, 128, 2048, BBALL, NGRP0);
    run_conv(stream, L[1], FB, Yar, meanb, varb, nullptr, FA, 128, 2048, BBALL, NGRP0);
    run_conv(stream, L[2], FA, Yar, meanb, varb, featF, nullptr, 0, 2048, BBALL, NGRP0);

    // ================= Few-shot head =================
    proto_sums_kernel<<<12 * 6, 128, 0, stream>>>(featF, sy, psums, pcnts, 128, 2048, 6);
    proto_combine_kernel<<<dim3(4, 6), 128, 0, stream>>>(psums, pcnts, proto, 128, 6, 3);
    euc_softmax_kernel<<<blk((long long)4 * 2048), 256, 0, stream>>>(
        proto, featF + (size_t)12 * 128 * 2048, (float*)d_out, 128, 2048, 6);
}
